// EnhancedGNNEncoder_22969485099217
// MI455X (gfx1250) — compile-verified, matrix operand order checked by reading, not run on hardware
//
#include <hip/hip_runtime.h>
#include <hip/hip_bf16.h>

typedef __attribute__((ext_vector_type(16))) _Float16 v16h;
typedef __attribute__((ext_vector_type(8)))  _Float16 v8h;
typedef __attribute__((ext_vector_type(4)))  _Float16 v4h;
typedef __attribute__((ext_vector_type(8)))  float    v8f;

// ---------------------------------------------------------------------------
// 16-bit fragment loader (ISA 7.12.2): each lane's 16 halves are two
// contiguous 8-half runs of a row:  [kc*32 + half*8 .. +7]  and  [+16 .. +23].
// Two ds_load_b128 per fragment.
// ---------------------------------------------------------------------------
__device__ __forceinline__ v16h load_frag(const _Float16* __restrict__ row,
                                          int kc, int half) {
    const v8h lo = *(const v8h*)(row + kc * 32 + half * 8);
    const v8h hi = *(const v8h*)(row + kc * 32 + 16 + half * 8);
    return __builtin_shufflevector(lo, hi, 0, 1, 2, 3, 4, 5, 6, 7,
                                   8, 9, 10, 11, 12, 13, 14, 15);
}

// ---------------------------------------------------------------------------
// Prep: Wn (L,NT,H,H)[d][o] f32 -> f16 transposed to [l][t][o][d]  (B^T layout
// so B fragments are K-contiguous); fc_w (OUT,H) is already [o][d].
// ---------------------------------------------------------------------------
__global__ void gnn_prep_kernel(const float* __restrict__ Wn,
                                const float* __restrict__ fc_w,
                                _Float16* __restrict__ WnHT,
                                _Float16* __restrict__ fcH) {
    const int i = blockIdx.x * 256 + threadIdx.x;
    const int nW = 2 * 3 * 64 * 64;
    if (i < nW) {
        const int lt = i >> 12;          // l*3+t
        const int o  = (i >> 6) & 63;
        const int d  = i & 63;
        WnHT[i] = (_Float16)Wn[(lt << 12) + (d << 6) + o];
    }
    if (i < 64 * 64) fcH[i] = (_Float16)fc_w[i];
}

// ---------------------------------------------------------------------------
// Edge phase: one thread per edge, float4 vector loads, HW f32 atomics.
// ---------------------------------------------------------------------------
__global__ void gnn_edge_kernel(const float* __restrict__ x,
                                const int*   __restrict__ ei,     // [2][E]
                                const float* __restrict__ eattr,  // [E][32]
                                const int*   __restrict__ etype,  // [E]
                                const float* __restrict__ Wm_l,   // [4][32]
                                const float* __restrict__ bm_l,   // [4]
                                const float* __restrict__ emb_l,  // [4][32]
                                float* __restrict__ aggr, int nE) {
    const int e = blockIdx.x * blockDim.x + threadIdx.x;
    if (e >= nE) return;
    const int src = ei[e];
    const int dst = ei[nE + e];
    const int t   = etype[e];
    const float4* ea4 = (const float4*)(eattr + (size_t)e * 32);
    const float4* wm4 = (const float4*)(Wm_l + t * 32);
    const float4* em4 = (const float4*)(emb_l + t * 32);
    float z = bm_l[t];
#pragma unroll
    for (int k = 0; k < 8; ++k) {
        const float4 a = ea4[k], b = em4[k], c = wm4[k];
        z += (a.x + b.x) * c.x + (a.y + b.y) * c.y +
             (a.z + b.z) * c.z + (a.w + b.w) * c.w;
    }
    const float w = (z > 20.f) ? z : log1pf(expf(z));   // stable softplus
    const float4* xs4 = (const float4*)(x + (size_t)src * 64);
    const float4* xd4 = (const float4*)(x + (size_t)dst * 64);
    float* ag = aggr + (size_t)dst * 64;
#pragma unroll
    for (int k = 0; k < 16; ++k) {
        const float4 s = xs4[k], d = xd4[k];
        unsafeAtomicAdd(&ag[k * 4 + 0], w * (s.x - d.x));
        unsafeAtomicAdd(&ag[k * 4 + 1], w * (s.y - d.y));
        unsafeAtomicAdd(&ag[k * 4 + 2], w * (s.z - d.z));
        unsafeAtomicAdd(&ag[k * 4 + 3], w * (s.w - d.w));
    }
}

// ---------------------------------------------------------------------------
// Node phase: per wave, one 16-node tile.  WMMA h_t = aggr_tile @ Wn[t] for
// all 3 node types, branch-free cndmask select, +bias, ReLU, register LN.
// ---------------------------------------------------------------------------
__global__ void __launch_bounds__(256)
gnn_node_kernel(const float* __restrict__ aggr,
                const int*   __restrict__ node_type,
                const _Float16* __restrict__ WnHT_l, // [3][64][64] f16 [o][d]
                const float* __restrict__ bn_l,      // [3][64]
                const float* __restrict__ ln_g,      // [64]
                const float* __restrict__ ln_b,      // [64]
                float* __restrict__ x_out, int nN) {
    __shared__ _Float16 sW[3 * 64 * 64];   // 24 KB: all 3 type matrices [o][d]
    __shared__ _Float16 sA[8][16 * 64];    // 16 KB: per-wave A tiles [m][d]

    const int tid = threadIdx.x;
    {   // vectorized weight stage: 1536 x v8h
        const v8h* wsrc = (const v8h*)WnHT_l;
        v8h* wdst = (v8h*)sW;
#pragma unroll
        for (int i = tid; i < 1536; i += 256) wdst[i] = wsrc[i];
    }
    __syncthreads();

    const int wave = tid >> 5, lane = tid & 31;
    const int base = (blockIdx.x * 8 + wave) * 16;
    if (base >= nN) return;

    _Float16* At = sA[wave];
    {   // stage A: each lane converts a contiguous 32-float run (node m=lane/2)
        const int m = lane >> 1, d0 = (lane & 1) * 32;
        const float4* src4 = (const float4*)(aggr + (size_t)(base + m) * 64 + d0);
        v4h* dst4 = (v4h*)(At + m * 64 + d0);
#pragma unroll
        for (int j = 0; j < 8; ++j) {
            const float4 f = src4[j];
            const v4h hv = { (_Float16)f.x, (_Float16)f.y,
                             (_Float16)f.z, (_Float16)f.w };
            dst4[j] = hv;
        }
    }
    __builtin_amdgcn_wave_barrier();

    const int h = lane >> 4, col0 = lane & 15;

    // hoisted A fragments (reused across 3 types x 4 col-tiles)
    const _Float16* arow = At + col0 * 64;           // M = lane & 15
    const v16h a0 = load_frag(arow, 0, h);
    const v16h a1 = load_frag(arow, 1, h);

    // ---- all 12 accumulators up front (pure WMMA phase) ----
    v8f acc[3][4];
#pragma unroll
    for (int ot = 0; ot < 4; ++ot) {
        const int bn_row = (ot * 16 + col0) * 64;    // N-row offset in [o][d]
#pragma unroll
        for (int t = 0; t < 3; ++t) {
            const _Float16* br = sW + t * 4096 + bn_row;
            v8f c = {};
            c = __builtin_amdgcn_wmma_f32_16x16x32_f16(false, a0, false, load_frag(br, 0, h), (short)0, c, false, false);
            c = __builtin_amdgcn_wmma_f32_16x16x32_f16(false, a1, false, load_frag(br, 1, h), (short)0, c, false, false);
            acc[t][ot] = c;
        }
    }

    // ---- hoisted per-col parameters (3 bias candidates + LN affine) ----
    float bia[3][4], lg[4], lb[4];
#pragma unroll
    for (int ot = 0; ot < 4; ++ot) {
        const int col = ot * 16 + col0;
        bia[0][ot] = bn_l[col];
        bia[1][ot] = bn_l[64 + col];
        bia[2][ot] = bn_l[128 + col];
        lg[ot] = ln_g[col];
        lb[ot] = ln_b[col];
    }

    // ---- vectorized node types: two b128 loads ----
    int tn[8];
    {
        const int4* np = (const int4*)(node_type + base + 8 * h);
        const int4 n0 = np[0], n1 = np[1];
        tn[0] = n0.x; tn[1] = n0.y; tn[2] = n0.z; tn[3] = n0.w;
        tn[4] = n1.x; tn[5] = n1.y; tn[6] = n1.z; tn[7] = n1.w;
    }

    // ---- branch-free select + bias + ReLU ----
    float vsel[4][8];
#pragma unroll
    for (int r = 0; r < 8; ++r) {
        const bool is0 = (tn[r] == 0), is1 = (tn[r] == 1);
#pragma unroll
        for (int ot = 0; ot < 4; ++ot) {
            const float hv = is0 ? acc[0][ot][r]
                           : is1 ? acc[1][ot][r]
                                 : acc[2][ot][r];
            const float bb = is0 ? bia[0][ot] : is1 ? bia[1][ot] : bia[2][ot];
            vsel[ot][r] = fmaxf(hv + bb, 0.f);       // ReLU
        }
    }

    // ---- LayerNorm: node (r + 8*h) spans 16 lanes of half-group x 4 tiles ----
#pragma unroll
    for (int r = 0; r < 8; ++r) {
        float s1 = 0.f, s2 = 0.f;
#pragma unroll
        for (int ot = 0; ot < 4; ++ot) { const float v = vsel[ot][r]; s1 += v; s2 += v * v; }
        for (int m = 1; m < 16; m <<= 1) {           // reduce within 16-lane half
            s1 += __shfl_xor(s1, m, 32);
            s2 += __shfl_xor(s2, m, 32);
        }
        const float mu   = s1 * (1.f / 64.f);
        const float var  = s2 * (1.f / 64.f) - mu * mu;
        const float rstd = rsqrtf(var + 1e-5f);
        const int node = base + r + 8 * h;
#pragma unroll
        for (int ot = 0; ot < 4; ++ot) {
            const int col = ot * 16 + col0;
            x_out[(size_t)node * 64 + col] =
                (vsel[ot][r] - mu) * rstd * lg[ot] + lb[ot];
        }
    }
}

// ---------------------------------------------------------------------------
// Final FC: out = x @ fc_w^T + fc_b via WMMA (fc_w native [o][d] == B^T layout)
// ---------------------------------------------------------------------------
__global__ void __launch_bounds__(256)
gnn_fc_kernel(const float* __restrict__ x,
              const _Float16* __restrict__ fcH,   // [64][64] f16 [o][d]
              const float* __restrict__ fc_b,     // [64]
              float* __restrict__ out, int nN) {
    __shared__ _Float16 sW[64 * 64];
    __shared__ _Float16 sA[8][16 * 64];

    const int tid = threadIdx.x;
    {
        const v8h* wsrc = (const v8h*)fcH;
        v8h* wdst = (v8h*)sW;
#pragma unroll
        for (int i = tid; i < 512; i += 256) wdst[i] = wsrc[i];
    }
    __syncthreads();

    const int wave = tid >> 5, lane = tid & 31;
    const int base = (blockIdx.x * 8 + wave) * 16;
    if (base >= nN) return;

    _Float16* At = sA[wave];
    {
        const int m = lane >> 1, d0 = (lane & 1) * 32;
        const float4* src4 = (const float4*)(x + (size_t)(base + m) * 64 + d0);
        v4h* dst4 = (v4h*)(At + m * 64 + d0);
#pragma unroll
        for (int j = 0; j < 8; ++j) {
            const float4 f = src4[j];
            const v4h hv = { (_Float16)f.x, (_Float16)f.y,
                             (_Float16)f.z, (_Float16)f.w };
            dst4[j] = hv;
        }
    }
    __builtin_amdgcn_wave_barrier();

    const int h = lane >> 4, col0 = lane & 15;
    const _Float16* arow = At + col0 * 64;
    const v16h a0 = load_frag(arow, 0, h);
    const v16h a1 = load_frag(arow, 1, h);

#pragma unroll
    for (int ot = 0; ot < 4; ++ot) {
        const _Float16* brow = sW + (ot * 16 + col0) * 64;
        v8f c = {};
        c = __builtin_amdgcn_wmma_f32_16x16x32_f16(false, a0, false, load_frag(brow, 0, h), (short)0, c, false, false);
        c = __builtin_amdgcn_wmma_f32_16x16x32_f16(false, a1, false, load_frag(brow, 1, h), (short)0, c, false, false);
        const int col = ot * 16 + col0;
        const float bias = fc_b[col];
#pragma unroll
        for (int r = 0; r < 8; ++r)
            out[(size_t)(base + r + 8 * h) * 64 + col] = c[r] + bias;
    }
}

// ---------------------------------------------------------------------------
extern "C" void kernel_launch(void* const* d_in, const int* in_sizes, int n_in,
                              void* d_out, int out_size, void* d_ws, size_t ws_size,
                              hipStream_t stream) {
    const float* x     = (const float*)d_in[0];
    const int*   ei    = (const int*)  d_in[1];
    const float* eattr = (const float*)d_in[2];
    const int*   ntyp  = (const int*)  d_in[3];
    const int*   etyp  = (const int*)  d_in[4];
    const float* Wn    = (const float*)d_in[5];
    const float* bn    = (const float*)d_in[6];
    const float* Wm    = (const float*)d_in[7];
    const float* bm    = (const float*)d_in[8];
    const float* emb   = (const float*)d_in[9];
    const float* ln_g  = (const float*)d_in[10];
    const float* ln_b  = (const float*)d_in[11];
    const float* fc_w  = (const float*)d_in[12];
    const float* fc_b  = (const float*)d_in[13];
    float* out = (float*)d_out;

    const int Nn = in_sizes[0] / 64;   // 100000
    const int Ne = in_sizes[2] / 32;   // 1600000

    // workspace layout (all 16B-aligned)
    float* aggr = (float*)d_ws;                            // N*64 f32
    float* xbuf = aggr + (size_t)Nn * 64;                  // N*64 f32
    _Float16* WnHT = (_Float16*)(xbuf + (size_t)Nn * 64);  // 2*3*4096 f16
    _Float16* fcH  = WnHT + 2 * 3 * 4096;                  // 4096 f16

    gnn_prep_kernel<<<96, 256, 0, stream>>>(Wn, fc_w, WnHT, fcH);

    const int edgeBlocks = (Ne + 255) / 256;
    const int tileBlocks = ((Nn + 15) / 16 + 7) / 8;

    for (int l = 0; l < 2; ++l) {
        hipMemsetAsync(aggr, 0, (size_t)Nn * 64 * sizeof(float), stream);
        const float* xin = (l == 0) ? x : xbuf;
        gnn_edge_kernel<<<edgeBlocks, 256, 0, stream>>>(
            xin, ei, eattr, etyp,
            Wm + l * 4 * 32, bm + l * 4, emb + l * 4 * 32, aggr, Ne);
        gnn_node_kernel<<<tileBlocks, 256, 0, stream>>>(
            aggr, ntyp, WnHT + l * 3 * 4096, bn + l * 3 * 64,
            ln_g + l * 64, ln_b + l * 64, xbuf, Nn);
    }
    gnn_fc_kernel<<<tileBlocks, 256, 0, stream>>>(xbuf, fcH, fc_b, out, Nn);
}
